// vSU2_41979010351316
// MI455X (gfx1250) — matrix operand order repaired and checked by reading, not compiled, not run on hardware
//
#include <hip/hip_runtime.h>
#include <hip/hip_bf16.h>
#include <stdint.h>

typedef __attribute__((ext_vector_type(16))) __bf16        v16bf;
typedef __attribute__((ext_vector_type(8)))  float         v8f;
typedef __attribute__((ext_vector_type(8)))  unsigned int  v8u;

#define NPTS  8192
#define NMASK 8191

__device__ __forceinline__ float elu1(float v) {
    return v > 0.0f ? v : expm1f(v);
}

// ---------------- small helper kernels ----------------

__global__ void k_zero(float* acc) {
    if (threadIdx.x < 4) acc[threadIdx.x] = 0.0f;
}

// sum(x2) and sum(exp(x2/(2*rl^2)))
__global__ void k_reg(const float* __restrict__ x, const float* __restrict__ rlen,
                      float* __restrict__ acc) {
    int i = blockIdx.x * blockDim.x + threadIdx.x;
    float a = x[i], b = x[NPTS + i], c = x[2*NPTS + i];
    float s = a*a + b*b + c*c;
    float inv = 1.0f / (2.0f * rlen[0] * rlen[0]);
    float e = expf(s * inv);
    for (int off = 16; off; off >>= 1) { s += __shfl_down(s, off); e += __shfl_down(e, off); }
    __shared__ float rs[8], re[8];
    int lane = threadIdx.x & 31, w = threadIdx.x >> 5;
    if (!lane) { rs[w] = s; re[w] = e; }
    __syncthreads();
    if (threadIdx.x == 0) {
        float S = 0.0f, E = 0.0f;
        for (int k = 0; k < 8; ++k) { S += rs[k]; E += re[k]; }
        atomicAdd(&acc[0], S);
        atomicAdd(&acc[1], E);
    }
}

// Pre-swizzle W1 (8192x64 f32, row-major fan_in x fan_out) into bf16 arranged as
// WMMA B fragments: for K-chunk kc and 16-col block cB, lane L holds 8 dwords
// (VGPR j = packed {K=2j+16*(L>=16), K+1} at column N=L&15), stored lane-contiguous:
//   dword index = ((kc*4 + cB)*32 + lane)*8 + j
__global__ void k_pack_w1(const float* __restrict__ W1, __hip_bfloat16* __restrict__ dst) {
    int idx = blockIdx.x * blockDim.x + threadIdx.x;   // 8192*64 elements
    int k  = idx >> 6, f = idx & 63;
    int kc = k >> 5,  kk = k & 31;
    int cB = f >> 4,  n  = f & 15;
    int hh = kk >> 4, j  = (kk & 15) >> 1, p = kk & 1;
    int lane = hh * 16 + n;
    int di = ((((kc*4 + cB) * 32 + lane) * 8) + j) * 2 + p;
    dst[di] = __float2bfloat16(W1[k*64 + f]);
}

// ---------------- fused corr + MLP kernel ----------------
// Block = 256 threads (8 waves), handles 32 rows of corr.
// Per K-chunk of 32: cooperative 32x32 corr tile (bf16, LDS) -> per-wave WMMA.
__global__ __launch_bounds__(256)
void k_main(const float* __restrict__ x, const __hip_bfloat16* __restrict__ wsB,
            const float* __restrict__ W2, const float* __restrict__ W3,
            float* __restrict__ acc) {
    extern __shared__ float smem[];
    float*          xs  = smem;                                  // 3*8192 f32 = 96 KB
    __hip_bfloat16* ct  = (__hip_bfloat16*)(smem + 3*NPTS);      // 32x32 bf16 = 2 KB
    unsigned int*   ctu = (unsigned int*)ct;
    float*          zbuf = smem + 3*NPTS + 512;                  // 32x64 f32 = 8 KB
    float*          z2t  = zbuf + 2048;                          // 64x32 f32 = 8 KB

    const int tid  = threadIdx.x;
    const int lane = tid & 31;
    const int wave = tid >> 5;
    const int hh   = lane >> 4;        // half-wave
    const int mA   = lane & 15;
    const int rb   = wave >> 2;        // row block (0..1)
    const int cb   = wave & 3;         // col block (0..3)
    const int i0   = blockIdx.x * 32;

    // stage x into LDS (stays resident)
    for (int idx4 = tid; idx4 < (3*NPTS)/4; idx4 += 256)
        ((float4*)xs)[idx4] = ((const float4*)x)[idx4];

    // corr-stage mapping: thread -> (row mC, 4 consecutive cols kC4)
    const int   mC   = tid >> 3;
    const int   kC4  = (tid & 7) * 4;
    const int   rowC = i0 + mC;
    const float xl0 = x[rowC], xl1 = x[NPTS + rowC], xl2 = x[2*NPTS + rowC];

    // A-fragment gather constants (contiguous A order used here, matching the W1 pack: VGPR j, lane L ->
    // K = (L>=16)*16 + j*2, pair {K,K+1}; row M = L&15)
    const int rowA16 = (16*rb + mA) * 16;   // row base in dword units (32 bf16/row)
    const int hh4    = hh * 4;
    const unsigned int* bbase = (const unsigned int*)wsB + (cb*32 + lane) * 8;

    v8f accv = {0.f,0.f,0.f,0.f,0.f,0.f,0.f,0.f};

    __syncthreads();   // xs ready

    for (int kc = 0; kc < NPTS/32; ++kc) {
        // ---- build 32x32 corr tile values (reads xs only) ----
        float s0, s1, s2, s3;
        {
            int b  = rowC + kc*32 + kC4;
            int a0 = (b    ) & NMASK, a1 = (b + 1) & NMASK;
            int a2 = (b + 2) & NMASK, a3 = (b + 3) & NMASK;
            s0 = xl0*xs[a0] + xl1*xs[NPTS+a0] + xl2*xs[2*NPTS+a0];
            s1 = xl0*xs[a1] + xl1*xs[NPTS+a1] + xl2*xs[2*NPTS+a1];
            s2 = xl0*xs[a2] + xl1*xs[NPTS+a2] + xl2*xs[2*NPTS+a2];
            s3 = xl0*xs[a3] + xl1*xs[NPTS+a3] + xl2*xs[2*NPTS+a3];
        }
        __syncthreads();   // prior fragment reads of ct complete
        {
            int cbase = mC*32 + kC4;
            ct[cbase+0] = __float2bfloat16(s0);
            ct[cbase+1] = __float2bfloat16(s1);
            ct[cbase+2] = __float2bfloat16(s2);
            ct[cbase+3] = __float2bfloat16(s3);
        }
        __syncthreads();   // tile visible to all waves

        // ---- per-wave fragments + WMMA ----
        v8u au;
        #pragma unroll
        for (int j = 0; j < 8; ++j)
            au[j] = ctu[rowA16 + hh4*2 + j];
        v8u bu = *(const v8u*)(bbase + kc*1024);

        v16bf A = __builtin_bit_cast(v16bf, au);
        v16bf B = __builtin_bit_cast(v16bf, bu);
        accv = __builtin_amdgcn_wmma_f32_16x16x32_bf16(
            false, A, false, B, (short)0, accv, false, false);
    }

    // h1 -> elu -> zbuf (C/D layout: VGPR v, lane L -> row = v + 8*(L>=16), col = L&15)
    #pragma unroll
    for (int v = 0; v < 8; ++v) {
        int row = 16*rb + 8*hh + v;
        zbuf[row*64 + 16*cb + mA] = elu1(accv[v]);
    }
    __syncthreads();

    // layer 2: z2t[f][r] = elu(sum_g zbuf[r][g] * W2[g][f])
    {
        int f  = tid & 63;
        int r0 = (tid >> 6) * 8;
        #pragma unroll
        for (int q = 0; q < 8; ++q) {
            int r = r0 + q;
            float s = 0.0f;
            for (int g = 0; g < 64; ++g) s += zbuf[r*64 + g] * W2[g*64 + f];
            z2t[f*32 + r] = elu1(s);
        }
    }
    __syncthreads();

    // layer 3 + row-sum reduce (wave 0)
    if (tid < 32) {
        float s = 0.0f;
        for (int f = 0; f < 64; ++f) s += z2t[f*32 + tid] * W3[f];
        float v = elu1(s);
        for (int off = 16; off; off >>= 1) v += __shfl_down(v, off);
        if (tid == 0) atomicAdd(&acc[2], v);
    }
}

__global__ void k_fin(const float* __restrict__ acc, const float* __restrict__ regu2,
                      const float* __restrict__ regu, float* __restrict__ out) {
    out[0] = acc[2] * (1.0f / (float)NPTS) - regu2[0]*acc[0] - regu[0]*acc[1];
}

// ---------------- launcher ----------------

extern "C" void kernel_launch(void* const* d_in, const int* in_sizes, int n_in,
                              void* d_out, int out_size, void* d_ws, size_t ws_size,
                              hipStream_t stream) {
    const float* x     = (const float*)d_in[0];
    const float* W1    = (const float*)d_in[1];
    const float* W2    = (const float*)d_in[2];
    const float* W3    = (const float*)d_in[3];
    const float* regu2 = (const float*)d_in[4];
    const float* regu  = (const float*)d_in[5];
    const float* rlen  = (const float*)d_in[6];
    float* out = (float*)d_out;

    float* acc = (float*)d_ws;                                       // 4 floats
    __hip_bfloat16* wsB = (__hip_bfloat16*)((char*)d_ws + 256);      // 1 MB bf16 W1 pack

    k_zero<<<1, 32, 0, stream>>>(acc);
    k_reg<<<NPTS/256, 256, 0, stream>>>(x, rlen, acc);
    k_pack_w1<<<(NPTS*64)/256, 256, 0, stream>>>(W1, wsB);

    size_t smem = (size_t)(3*NPTS + 512 + 2048 + 2048) * sizeof(float); // 116736 B
    k_main<<<NPTS/32, 256, smem, stream>>>(x, wsB, W2, W3, acc);

    k_fin<<<1, 1, 0, stream>>>(acc, regu2, regu, out);
}
